// AttentionBlock_29291676959363
// MI455X (gfx1250) — matrix operand order repaired and checked by reading, not compiled, not run on hardware
//
#include <hip/hip_runtime.h>

typedef __attribute__((ext_vector_type(16))) __bf16          v16bf;
typedef __attribute__((ext_vector_type(8)))  float           v8f;
typedef __attribute__((ext_vector_type(8)))  unsigned short  v8us;
typedef __attribute__((ext_vector_type(16))) unsigned short  v16us;

union BF16x16 { v16us u; v16bf b; };

#define S_LEN 2048
#define D_DIM 512
#define BR    64
#define BC    32

__device__ __forceinline__ unsigned short f2bf(float f) {
  // round-to-nearest-even f32 -> bf16
  unsigned int u = __float_as_uint(f);
  u += 0x7FFFu + ((u >> 16) & 1u);
  return (unsigned short)(u >> 16);
}

__global__ __launch_bounds__(256)
void fattn_cdna5_kernel(const float* __restrict__ Kg,
                        const float* __restrict__ Qg,
                        const float* __restrict__ Vg,
                        const float* __restrict__ Og,
                        float* __restrict__ out)
{
  extern __shared__ unsigned char smem_raw[];
  unsigned short* sQ  = (unsigned short*)smem_raw;        // 64 x 512 bf16
  unsigned short* sK  = sQ  + BR * D_DIM;                 // 32 x 512 bf16
  unsigned short* sVT = sK  + BC * D_DIM;                 // 512 x 32 bf16 (transposed)
  float*          sS  = (float*)(sVT + D_DIM * BC);       // 64 x 32 logits
  unsigned short* sP  = (unsigned short*)(sS + BR * BC);  // 64 x 32 probs bf16
  float*          sM  = (float*)(sP + BR * BC);           // running max
  float*          sL  = sM + BR;                          // running sum
  float*          sA  = sL + BR;                          // per-block alpha

  const int tid   = threadIdx.x;
  const int b     = blockIdx.y;
  const int qbase = blockIdx.x * BR;
  const int w     = tid >> 5;
  const int lane  = tid & 31;
  const int lm    = lane & 15;   // N / M index within subtile
  const int kh    = lane >> 4;   // lane half
  const int rg    = w & 3;       // query row-group (16 rows each)
  const int nh    = w >> 2;      // QK: key half | PV: column half

  const size_t bso = (size_t)b * S_LEN * D_DIM;

  // ---- copy `original` to left half of output ----
  for (int i = tid; i < BR * D_DIM / 4; i += 256) {
    int row = (i * 4) / D_DIM;
    int dd  = (i * 4) % D_DIM;
    float4 v = *(const float4*)&Og[bso + (size_t)(qbase + row) * D_DIM + dd];
    *(float4*)&out[((size_t)b * S_LEN + qbase + row) * (2 * D_DIM) + dd] = v;
  }

  // ---- stage Q tile to LDS (f32 -> bf16) ----
  for (int i = tid; i < BR * D_DIM / 4; i += 256) {
    int row = (i * 4) / D_DIM;
    int dd  = (i * 4) % D_DIM;
    float4 v = *(const float4*)&Qg[bso + (size_t)(qbase + row) * D_DIM + dd];
    unsigned int w0 = (unsigned int)f2bf(v.x) | ((unsigned int)f2bf(v.y) << 16);
    unsigned int w1 = (unsigned int)f2bf(v.z) | ((unsigned int)f2bf(v.w) << 16);
    *(unsigned int*)&sQ[row * D_DIM + dd]     = w0;
    *(unsigned int*)&sQ[row * D_DIM + dd + 2] = w1;
  }

  if (tid < BR) { sM[tid] = -1e30f; sL[tid] = 0.f; sA[tid] = 0.f; }

  v8f acc[16] = {};   // 16 rows x 256 cols of O, f32

  __syncthreads();

  const int nblocks = qbase / BC + 2;  // causal: last block touches qbase+63
  for (int jb = 0; jb < nblocks; ++jb) {
    const int jbase = jb * BC;

    // ---- stage K (row-major) and V (transposed) tiles, f32 -> bf16 ----
    for (int i = tid; i < BC * D_DIM / 4; i += 256) {
      int key = (i * 4) / D_DIM;
      int dd  = (i * 4) % D_DIM;
      size_t g = bso + (size_t)(jbase + key) * D_DIM + dd;
      float4 kv = *(const float4*)&Kg[g];
      unsigned int w0 = (unsigned int)f2bf(kv.x) | ((unsigned int)f2bf(kv.y) << 16);
      unsigned int w1 = (unsigned int)f2bf(kv.z) | ((unsigned int)f2bf(kv.w) << 16);
      *(unsigned int*)&sK[key * D_DIM + dd]     = w0;
      *(unsigned int*)&sK[key * D_DIM + dd + 2] = w1;
      float4 vv = *(const float4*)&Vg[g];
      sVT[(dd + 0) * BC + key] = f2bf(vv.x);
      sVT[(dd + 1) * BC + key] = f2bf(vv.y);
      sVT[(dd + 2) * BC + key] = f2bf(vv.z);
      sVT[(dd + 3) * BC + key] = f2bf(vv.w);
    }
    __syncthreads();

    // ---- QK^T: each wave -> one 16x16 subtile of the 64x32 logits ----
    {
      v8f c = {};
      const unsigned short* qrow = &sQ[(rg * 16 + lm) * D_DIM];
      const unsigned short* krow = &sK[(nh * 16 + lm) * D_DIM];
      #pragma unroll
      for (int kc = 0; kc < D_DIM / 32; ++kc) {
        BF16x16 a, bm;
        a.u  = *(const v16us*)&qrow[kc * 32 + kh * 16];        // contiguous 16 dims
        bm.u = *(const v16us*)&krow[kc * 32 + kh * 16];        // contiguous 16 dims
        c = __builtin_amdgcn_wmma_f32_16x16x32_bf16(false, a.b, false, bm.b,
                                                    (short)0, c, false, false);
      }
      #pragma unroll
      for (int r = 0; r < 8; ++r)
        sS[(rg * 16 + r + 8 * kh) * BC + nh * 16 + lm] = c[r];
    }
    __syncthreads();

    // ---- online softmax: one thread per query row ----
    if (tid < BR) {
      int qr = qbase + tid;
      float m_old = sM[tid];
      float mx = m_old;
      #pragma unroll
      for (int c = 0; c < BC; ++c) {
        float sv = sS[tid * BC + c];
        if ((jbase + c) <= qr && sv > mx) mx = sv;
      }
      float al  = __expf(m_old - mx);
      float sum = 0.f;
      #pragma unroll
      for (int c = 0; c < BC; ++c) {
        float pv = ((jbase + c) <= qr) ? __expf(sS[tid * BC + c] - mx) : 0.f;
        sP[tid * BC + c] = f2bf(pv);
        sum += pv;
      }
      sM[tid] = mx;
      sL[tid] = al * sL[tid] + sum;
      sA[tid] = al;
    }
    __syncthreads();

    // ---- PV: rescale accumulator by alpha, then accumulate P x V ----
    {
      float al[8];
      #pragma unroll
      for (int r = 0; r < 8; ++r) al[r] = sA[rg * 16 + r + 8 * kh];
      #pragma unroll
      for (int cs = 0; cs < 16; ++cs)
        #pragma unroll
        for (int r = 0; r < 8; ++r) acc[cs][r] *= al[r];

      BF16x16 pa;
      const unsigned short* prow = &sP[(rg * 16 + lm) * BC];
      pa.u = *(const v16us*)&prow[kh * 16];

      #pragma unroll
      for (int cs = 0; cs < 16; ++cs) {
        int dim = nh * 256 + cs * 16 + lm;
        BF16x16 vb;
        vb.u = *(const v16us*)&sVT[dim * BC + kh * 16];  // 16 contiguous keys
        acc[cs] = __builtin_amdgcn_wmma_f32_16x16x32_bf16(false, pa.b, false, vb.b,
                                                          (short)0, acc[cs], false, false);
      }
    }
    __syncthreads();  // protect sK/sVT/sP before next block's staging
  }

  // ---- normalize and store right half of output ----
  float linv[8];
  #pragma unroll
  for (int r = 0; r < 8; ++r) linv[r] = 1.f / sL[rg * 16 + r + 8 * kh];
  #pragma unroll
  for (int cs = 0; cs < 16; ++cs) {
    int col = D_DIM + nh * 256 + cs * 16 + lm;
    #pragma unroll
    for (int r = 0; r < 8; ++r) {
      int row = qbase + rg * 16 + r + 8 * kh;
      out[((size_t)b * S_LEN + row) * (2 * D_DIM) + col] = acc[cs][r] * linv[r];
    }
  }
}

extern "C" void kernel_launch(void* const* d_in, const int* in_sizes, int n_in,
                              void* d_out, int out_size, void* d_ws, size_t ws_size,
                              hipStream_t stream) {
  // setup_inputs order: keys, queries, values, original
  const float* K = (const float*)d_in[0];
  const float* Q = (const float*)d_in[1];
  const float* V = (const float*)d_in[2];
  const float* O = (const float*)d_in[3];
  float* out = (float*)d_out;

  const int B = 4;
  dim3 grid(S_LEN / BR, B);
  dim3 block(256);
  size_t shmem =
      (size_t)(BR * D_DIM + BC * D_DIM + D_DIM * BC) * sizeof(unsigned short) // Q,K,V^T
      + (size_t)BR * BC * sizeof(float)            // logits
      + (size_t)BR * BC * sizeof(unsigned short)   // probs
      + 3u * BR * sizeof(float);                   // m, l, alpha

  hipLaunchKernelGGL(fattn_cdna5_kernel, grid, block, shmem, stream, K, Q, V, O, out);
}
